// EGNN_dynamics_QM9_MC_76364518523001
// MI455X (gfx1250) — matrix operand order repaired and checked by reading, not compiled, not run on hardware
//
#include <hip/hip_runtime.h>
#include <hip/hip_bf16.h>
#include <math.h>

typedef _Float16 h16_t;
typedef __attribute__((ext_vector_type(16))) _Float16 v16h;
typedef __attribute__((ext_vector_type(8)))  float    v8f;

#define BB 16
#define NN 64
#define MM (BB*NN)            // 1024 nodes
#define HID 256
#define KE_PAD 544            // 512 + vi (<=519) padded to multiple of 32
#define KN 768                // node MLP input
#define PHI_PAD 64            // vi*vo (<=49) padded
#define VSTRIDE 8             // coord vector channel stride (vi<=7)
#define CSTRIDE 24            // 3 * VSTRIDE floats per node
#define NODE_CHUNK 256        // nodes per edge-chunk
#define ECH (NODE_CHUNK*NN)   // 16384 edges per chunk
#define NCHUNKS (MM/NODE_CHUNK)
#define EPSV 1e-10f

// ---------------------------------------------------------------------------
// Generic WMMA GEMM: out = epi(A[Mr,K] @ Wp + bias)
//  Wp pre-permuted to B-fragment layout ( (kt*ntiles+nt)*32 + lane )*16 halves.
//  One wave = 32(M) x 64(N) output tile: 8 WMMA per k-step, each B fragment
//  reused by two A strips (halves weight-fragment traffic vs 16x64).
//  Epilogue: +bias -> optional SiLU -> optional +resid -> optional *rowscale.
// ---------------------------------------------------------------------------
__global__ __launch_bounds__(256) void gemm_wmma_kernel(
    const h16_t* __restrict__ A, const h16_t* __restrict__ Wp,
    const float* __restrict__ bias, int nreal,
    const float* __restrict__ rowscale, const float* __restrict__ resid,
    h16_t* __restrict__ out16, float* __restrict__ out32,
    int Mr, int K, int Np, int act_silu)
{
    const int lane = threadIdx.x & 31;
    const int wave = blockIdx.x * (blockDim.x >> 5) + (threadIdx.x >> 5);
    const int mstrips = Mr >> 5;               // 32 rows per wave
    const int ngroups = Np >> 6;
    if (wave >= mstrips * ngroups) return;
    const int ms    = wave % mstrips;
    const int ng    = wave / mstrips;
    const int mbase = ms << 5;
    const int row0  = mbase + (lane & 15);     // A strip 0
    const int row1  = row0 + 16;               // A strip 1
    const int kh    = (lane < 16) ? 0 : 16;    // A-fragment K sub-offset
    const int ntiles = Np >> 4;
    const int nt0    = ng << 2;

    v8f acc[2][4] = {};

    const int ksteps = K >> 5;
    for (int kt = 0; kt < ksteps; ++kt) {
        union { v16h v; uint4 q[2]; } a0, a1;
        const h16_t* ap0 = A + (size_t)row0 * K + (kt << 5) + kh;
        const h16_t* ap1 = A + (size_t)row1 * K + (kt << 5) + kh;
        a0.q[0] = *(const uint4*)ap0;          // K[kh .. kh+8)
        a0.q[1] = *(const uint4*)(ap0 + 8);    // K[kh+8 .. kh+16)
        a1.q[0] = *(const uint4*)ap1;
        a1.q[1] = *(const uint4*)(ap1 + 8);
#pragma unroll
        for (int t = 0; t < 4; ++t) {
            union { v16h v; uint4 q[2]; } b;
            const h16_t* bp = Wp + (((size_t)kt * ntiles + (nt0 + t)) * 32 + lane) * 16;
            b.q[0] = *(const uint4*)bp;
            b.q[1] = *(const uint4*)(bp + 8);
            acc[0][t] = __builtin_amdgcn_wmma_f32_16x16x32_f16(
                            false, a0.v, false, b.v, (short)0, acc[0][t], false, false);
            acc[1][t] = __builtin_amdgcn_wmma_f32_16x16x32_f16(
                            false, a1.v, false, b.v, (short)0, acc[1][t], false, false);
        }
    }

#pragma unroll
    for (int hlf = 0; hlf < 2; ++hlf) {
#pragma unroll
        for (int t = 0; t < 4; ++t) {
            const int col = ((nt0 + t) << 4) + (lane & 15);
            const float bv = (bias && col < nreal) ? bias[col] : 0.f;
#pragma unroll
            for (int r = 0; r < 8; ++r) {
                const int rr = mbase + (hlf << 4) + ((lane < 16) ? r : (8 + r));
                float v = acc[hlf][t][r] + bv;
                if (act_silu) v = v * (1.f / (1.f + __expf(-v)));
                if (resid)    v += resid[(size_t)rr * Np + col];
                if (rowscale) v *= rowscale[rr];
                const size_t oi = (size_t)rr * Np + col;
                if (out32) out32[oi] = v;
                if (out16) out16[oi] = (h16_t)v;
            }
        }
    }
}

// ---------------------------------------------------------------------------
// Permute f32 weight [K,Nreal] -> f16 B-fragment tiles [Kpad/32][Npad/16][32][16]
// lane 0-15: K = kt*32 + 0..15, col = nt*16+lane; lane 16-31: K = kt*32 + 16..31
// ---------------------------------------------------------------------------
__global__ __launch_bounds__(256) void convert_weight_kernel(
    const float* __restrict__ W, h16_t* __restrict__ Wp,
    int K, int Nreal, int Kpad, int Npad)
{
    const int lane = threadIdx.x & 31;
    const int tile = blockIdx.x * (blockDim.x >> 5) + (threadIdx.x >> 5);
    const int ntiles = Npad >> 4;
    const int ktiles = Kpad >> 5;
    if (tile >= ktiles * ntiles) return;
    const int nt = tile % ntiles;
    const int kt = tile / ntiles;
    const int nn = (nt << 4) + (lane & 15);
    const int kb = (kt << 5) + ((lane < 16) ? 0 : 16);
    union { h16_t h[16]; uint4 q[2]; } tmp;
#pragma unroll
    for (int i = 0; i < 16; ++i) {
        const int kk = kb + i;
        float v = (kk < K && nn < Nreal) ? W[(size_t)kk * Nreal + nn] : 0.f;
        tmp.h[i] = (h16_t)v;
    }
    h16_t* dst = Wp + ((size_t)tile * 32 + lane) * 16;
    *(uint4*)dst       = tmp.q[0];
    *(uint4*)(dst + 8) = tmp.q[1];
}

// ---------------------------------------------------------------------------
// Embedding: coord init, h = (masked_feats @ W + b) * nm  (K=16, trivial)
// ---------------------------------------------------------------------------
__global__ void embed_kernel(const float* __restrict__ xh, const float* __restrict__ nmk,
                             const float* __restrict__ w, const float* __restrict__ bias,
                             float* __restrict__ coord, float* __restrict__ hF,
                             h16_t* __restrict__ h16, h16_t* __restrict__ h0)
{
    const int bi = blockIdx.x;
    const int c  = threadIdx.x;          // 256
    const float nm = nmk[bi];
    if (c < 3) coord[(size_t)bi * CSTRIDE + c * VSTRIDE] = xh[bi * 19 + c] * nm;
    float s = bias[c];
    for (int k = 0; k < 16; ++k) s += (xh[bi * 19 + 3 + k] * nm) * w[k * HID + c];
    s *= nm;
    hF [(size_t)bi * HID + c] = s;
    h16[(size_t)bi * HID + c] = (h16_t)s;
    h0 [(size_t)bi * HID + c] = (h16_t)s;
}

// ---------------------------------------------------------------------------
// Edge input: Xe row = [h_i | h_j | radial | 0-pad], also stores diff.
// One wave per edge; 8 edges per 256-thread block.
// ---------------------------------------------------------------------------
__global__ __launch_bounds__(256) void edge_input_kernel(
    const h16_t* __restrict__ h16, const float* __restrict__ coord,
    float* __restrict__ Dbuf, h16_t* __restrict__ Xe, int e0, int vi)
{
    const int lane = threadIdx.x & 31;
    const int el = blockIdx.x * 8 + (threadIdx.x >> 5);    // chunk-local edge
    const int eg = e0 + el;
    const int gi = eg / NN;                                // global node i
    const int j  = eg - gi * NN;
    const int gj = (gi / NN) * NN + j;                     // global node j
    h16_t* dst = Xe + (size_t)el * KE_PAD;
    *(uint4*)(dst + lane * 8)       = *(const uint4*)(h16 + (size_t)gi * HID + lane * 8);
    *(uint4*)(dst + HID + lane * 8) = *(const uint4*)(h16 + (size_t)gj * HID + lane * 8);
    float rad = 0.f;
    if (lane < vi) {
#pragma unroll
        for (int d = 0; d < 3; ++d) {
            float df = coord[(size_t)gi * CSTRIDE + d * VSTRIDE + lane]
                     - coord[(size_t)gj * CSTRIDE + d * VSTRIDE + lane];
            Dbuf[(size_t)el * CSTRIDE + d * VSTRIDE + lane] = df;
            rad += df * df;
        }
    }
    dst[2 * HID + lane] = (lane < vi) ? (h16_t)rad : (h16_t)0.f;  // cols 512..543
}

// ---------------------------------------------------------------------------
// agg = sum_j m, and build node-MLP input Xn = [h | agg | node_attr]
// ---------------------------------------------------------------------------
__global__ void agg_kernel(const h16_t* __restrict__ m16, const h16_t* __restrict__ h16,
                           const h16_t* __restrict__ h0, h16_t* __restrict__ Xn, int bi0)
{
    const int bl = blockIdx.x;
    const int bi = bi0 + bl;
    const int c  = threadIdx.x;          // 256
    float s = 0.f;
    for (int j = 0; j < NN; ++j) s += (float)m16[((size_t)bl * NN + j) * HID + c];
    Xn[(size_t)bi * KN + c]           = h16[(size_t)bi * HID + c];
    Xn[(size_t)bi * KN + HID + c]     = (h16_t)s;
    Xn[(size_t)bi * KN + 2 * HID + c] = h0[(size_t)bi * HID + c];
}

// ---------------------------------------------------------------------------
// trans einsum + coord mix update
// ---------------------------------------------------------------------------
__global__ void trans_kernel(const float* __restrict__ Dbuf, const float* __restrict__ phi,
                             const float* __restrict__ cIn, const float* __restrict__ mix,
                             const float* __restrict__ nmk, float* __restrict__ cOut,
                             int bi0, int vi, int vo)
{
    const int bl = blockIdx.x;
    const int bi = bi0 + bl;
    const int t  = threadIdx.x;          // 32, use 3*vo
    if (t >= 3 * vo) return;
    const int d = t / vo, w = t - d * vo;
    float s = 0.f;
    for (int j = 0; j < NN; ++j) {
        const size_t e = (size_t)bl * NN + j;
        for (int v = 0; v < vi; ++v)
            s += Dbuf[e * CSTRIDE + d * VSTRIDE + v] * phi[e * PHI_PAD + v * vo + w];
    }
    s *= (1.f / NN);
    float cs = 0.f;
    for (int v = 0; v < vi; ++v)
        cs += cIn[(size_t)bi * CSTRIDE + d * VSTRIDE + v] * mix[v * vo + w];
    cOut[(size_t)bi * CSTRIDE + d * VSTRIDE + w] = (cs + s) * nmk[bi];
}

// ---------------------------------------------------------------------------
// h_final = (h @ emb_out_w + b) * nm    (N=16, trivial)
// ---------------------------------------------------------------------------
__global__ void hfinal_kernel(const float* __restrict__ hF, const float* __restrict__ w,
                              const float* __restrict__ bias, const float* __restrict__ nmk,
                              float* __restrict__ hfin)
{
    const int bi = blockIdx.x;
    const int c  = threadIdx.x;          // 32
    if (c >= 16) return;
    float s = bias[c];
    for (int k = 0; k < HID; ++k) s += hF[(size_t)bi * HID + k] * w[k * 16 + c];
    hfin[bi * 16 + c] = s * nmk[bi];
}

// ---------------------------------------------------------------------------
// Gram-Schmidt frame, 3x3 inverse, mean-centering, output assembly.
// One block per batch (64 threads = nodes), LDS tree reduction for the mean.
// ---------------------------------------------------------------------------
__global__ void geom_kernel(const float* __restrict__ coord, const float* __restrict__ xh,
                            const float* __restrict__ nmk, const float* __restrict__ hfin,
                            float* __restrict__ out)
{
    __shared__ float sx[3][NN];
    __shared__ float sn[NN];
    const int b  = blockIdx.x;
    const int i  = threadIdx.x;          // 64
    const int bi = b * NN + i;
    const float nm = nmk[bi];
    float v1[3], v2[3];
#pragma unroll
    for (int d = 0; d < 3; ++d) {
        v1[d] = coord[(size_t)bi * CSTRIDE + d * VSTRIDE + 0] * nm;
        v2[d] = coord[(size_t)bi * CSTRIDE + d * VSTRIDE + 1] * nm;
    }
    float l1 = sqrtf(v1[0]*v1[0] + v1[1]*v1[1] + v1[2]*v1[2]) + EPSV;
    float n1[3] = { v1[0]/l1, v1[1]/l1, v1[2]/l1 };
    float dp = n1[0]*v2[0] + n1[1]*v2[1] + n1[2]*v2[2];
    float p2[3] = { v2[0]-dp*n1[0], v2[1]-dp*n1[1], v2[2]-dp*n1[2] };
    float l2 = sqrtf(p2[0]*p2[0] + p2[1]*p2[1] + p2[2]*p2[2]) + EPSV;
    float n2[3] = { p2[0]/l2, p2[1]/l2, p2[2]/l2 };
    float n3[3] = { n1[1]*n2[2]-n1[2]*n2[1], n1[2]*n2[0]-n1[0]*n2[2], n1[0]*n2[1]-n1[1]*n2[0] };
    float Q[3][3];
#pragma unroll
    for (int d = 0; d < 3; ++d) { Q[0][d]=n1[d]; Q[1][d]=n2[d]; Q[2][d]=n3[d]; }
    float A[3][3];
#pragma unroll
    for (int r = 0; r < 3; ++r)
#pragma unroll
        for (int c = 0; c < 3; ++c) A[r][c] = Q[r][c] + ((r==c) ? 1e-16f : 0.f);
    float c00 =  (A[1][1]*A[2][2]-A[1][2]*A[2][1]);
    float c01 = -(A[1][0]*A[2][2]-A[1][2]*A[2][0]);
    float c02 =  (A[1][0]*A[2][1]-A[1][1]*A[2][0]);
    float det = A[0][0]*c00 + A[0][1]*c01 + A[0][2]*c02;
    float id  = 1.f / det;
    float inv[3][3];
    inv[0][0] = c00*id;
    inv[1][0] = c01*id;
    inv[2][0] = c02*id;
    inv[0][1] = -(A[0][1]*A[2][2]-A[0][2]*A[2][1])*id;
    inv[1][1] =  (A[0][0]*A[2][2]-A[0][2]*A[2][0])*id;
    inv[2][1] = -(A[0][0]*A[2][1]-A[0][1]*A[2][0])*id;
    inv[0][2] =  (A[0][1]*A[1][2]-A[0][2]*A[1][1])*id;
    inv[1][2] = -(A[0][0]*A[1][2]-A[0][2]*A[1][0])*id;
    inv[2][2] =  (A[0][0]*A[1][1]-A[0][1]*A[1][0])*id;
    float x[3];
#pragma unroll
    for (int d = 0; d < 3; ++d) x[d] = xh[bi * 19 + d] * nm;
    float xi[3];
#pragma unroll
    for (int jj = 0; jj < 3; ++jj)
        xi[jj] = (x[0]*inv[0][jj] + x[1]*inv[1][jj] + x[2]*inv[2][jj]) * nm;
    sx[0][i] = xi[0] * nm; sx[1][i] = xi[1] * nm; sx[2][i] = xi[2] * nm; sn[i] = nm;
    __syncthreads();
    for (int s2 = 32; s2 > 0; s2 >>= 1) {
        if (i < s2) {
            sx[0][i] += sx[0][i+s2]; sx[1][i] += sx[1][i+s2];
            sx[2][i] += sx[2][i+s2]; sn[i]    += sn[i+s2];
        }
        __syncthreads();
    }
    const float isn = 1.f / sn[0];
    float* o = out + (size_t)bi * 19;
    o[0] = (xi[0] - sx[0][0]*isn) * nm;
    o[1] = (xi[1] - sx[1][0]*isn) * nm;
    o[2] = (xi[2] - sx[2][0]*isn) * nm;
    for (int c = 0; c < 16; ++c) o[3 + c] = hfin[bi * 16 + c];
    float* oi = out + (size_t)MM * 19 + (size_t)bi * 9;
    float* oq = out + (size_t)MM * 19 + (size_t)MM * 9 + (size_t)bi * 9;
#pragma unroll
    for (int r = 0; r < 3; ++r)
#pragma unroll
        for (int c = 0; c < 3; ++c) { oi[r*3+c] = inv[r][c]; oq[r*3+c] = Q[r][c]; }
}

// ---------------------------------------------------------------------------
extern "C" void kernel_launch(void* const* d_in, const int* in_sizes, int n_in,
                              void* d_out, int out_size, void* d_ws, size_t ws_size,
                              hipStream_t stream)
{
    (void)in_sizes; (void)n_in; (void)out_size; (void)ws_size;
    const float* xh  = (const float*)d_in[0];
    const float* nmk = (const float*)d_in[1];
    const float* em  = (const float*)d_in[2];
    const float* emb_in_b  = (const float*)d_in[3];
    const float* emb_in_w  = (const float*)d_in[4];
    const float* emb_out_b = (const float*)d_in[5];
    const float* emb_out_w = (const float*)d_in[6];
    // per-layer leaves (sorted keys): 0 coord_b1, 1 coord_b2, 2 coord_w1, 3 coord_w2,
    // 4 edge_b1, 5 edge_b2, 6 edge_w1, 7 edge_w2, 8 mix, 9 node_b1, 10 node_b2,
    // 11 node_w1, 12 node_w2
    auto LP = [&](int L, int k) { return (const float*)d_in[7 + 13 * L + k]; };

    char* base = (char*)d_ws;
    size_t off = 0;
    auto alloc = [&](size_t bytes) -> void* {
        off = (off + 255) & ~(size_t)255;
        void* p = base + off; off += bytes; return p;
    };
    float* coordA = (float*)alloc((size_t)MM * CSTRIDE * 4);
    float* coordB = (float*)alloc((size_t)MM * CSTRIDE * 4);
    float* hF     = (float*)alloc((size_t)MM * HID * 4);
    h16_t* h16b   = (h16_t*)alloc((size_t)MM * HID * 2);
    h16_t* h0b    = (h16_t*)alloc((size_t)MM * HID * 2);
    h16_t* Xn     = (h16_t*)alloc((size_t)MM * KN * 2);
    h16_t* u1     = (h16_t*)alloc((size_t)MM * HID * 2);
    float* hfin   = (float*)alloc((size_t)MM * 16 * 4);
    h16_t* Xe     = (h16_t*)alloc((size_t)ECH * KE_PAD * 2);
    h16_t* m1     = (h16_t*)alloc((size_t)ECH * HID * 2);   // also reused as p1
    h16_t* m16b   = (h16_t*)alloc((size_t)ECH * HID * 2);
    float* phi    = (float*)alloc((size_t)ECH * PHI_PAD * 4);
    float* Dbuf   = (float*)alloc((size_t)ECH * CSTRIDE * 4);
    h16_t* WpE1   = (h16_t*)alloc((size_t)KE_PAD * HID * 2);
    h16_t* WpE2   = (h16_t*)alloc((size_t)HID * HID * 2);
    h16_t* WpC1   = (h16_t*)alloc((size_t)HID * HID * 2);
    h16_t* WpC2   = (h16_t*)alloc((size_t)HID * PHI_PAD * 2);
    h16_t* WpN1   = (h16_t*)alloc((size_t)KN * HID * 2);
    h16_t* WpN2   = (h16_t*)alloc((size_t)HID * HID * 2);

    auto launch_conv = [&](const float* W, h16_t* Wp, int K, int Nreal, int Kpad, int Npad) {
        int tiles = (Kpad / 32) * (Npad / 16);
        convert_weight_kernel<<<(tiles + 7) / 8, 256, 0, stream>>>(W, Wp, K, Nreal, Kpad, Npad);
    };
    auto launch_gemm = [&](const h16_t* A, const h16_t* Wp, const float* bias, int nreal,
                           const float* rowscale, const float* resid,
                           h16_t* o16, float* o32, int Mr, int K, int Np, int silu) {
        int total = (Mr / 32) * (Np / 64);
        gemm_wmma_kernel<<<(total + 7) / 8, 256, 0, stream>>>(
            A, Wp, bias, nreal, rowscale, resid, o16, o32, Mr, K, Np, silu);
    };

    embed_kernel<<<MM, HID, 0, stream>>>(xh, nmk, emb_in_w, emb_in_b, coordA, hF, h16b, h0b);

    const int vins[4]  = {1, 7, 7, 7};
    const int vouts[4] = {7, 7, 7, 3};
    float* cIn = coordA;
    float* cOut = coordB;
    for (int L = 0; L < 4; ++L) {
        const int vi = vins[L], vo = vouts[L];
        launch_conv(LP(L, 6),  WpE1, 512 + vi, HID,     KE_PAD, HID);
        launch_conv(LP(L, 7),  WpE2, HID,      HID,     HID,    HID);
        launch_conv(LP(L, 2),  WpC1, HID,      HID,     HID,    HID);
        launch_conv(LP(L, 3),  WpC2, HID,      vi * vo, HID,    PHI_PAD);
        launch_conv(LP(L, 11), WpN1, KN,       HID,     KN,     HID);
        launch_conv(LP(L, 12), WpN2, HID,      HID,     HID,    HID);
        for (int ch = 0; ch < NCHUNKS; ++ch) {
            const int e0  = ch * ECH;
            const int bi0 = ch * NODE_CHUNK;
            edge_input_kernel<<<ECH / 8, 256, 0, stream>>>(h16b, cIn, Dbuf, Xe, e0, vi);
            // m1 = silu(Xe @ edge_w1 + b1)
            launch_gemm(Xe, WpE1, LP(L, 4), HID, nullptr, nullptr, m1, nullptr, ECH, KE_PAD, HID, 1);
            // m = silu(m1 @ edge_w2 + b2) * edge_mask
            launch_gemm(m1, WpE2, LP(L, 5), HID, em + e0, nullptr, m16b, nullptr, ECH, HID, HID, 1);
            agg_kernel<<<NODE_CHUNK, HID, 0, stream>>>(m16b, h16b, h0b, Xn, bi0);
            // p1 = silu(m @ coord_w1 + c1)   (reuse m1 buffer)
            launch_gemm(m16b, WpC1, LP(L, 0), HID, nullptr, nullptr, m1, nullptr, ECH, HID, HID, 1);
            // phi = (p1 @ coord_w2 + c2) * edge_mask
            launch_gemm(m1, WpC2, LP(L, 1), vi * vo, em + e0, nullptr, nullptr, phi, ECH, HID, PHI_PAD, 0);
            trans_kernel<<<NODE_CHUNK, 32, 0, stream>>>(Dbuf, phi, cIn, LP(L, 8), nmk, cOut, bi0, vi, vo);
        }
        // u1 = silu(Xn @ node_w1 + nb1)
        launch_gemm(Xn, WpN1, LP(L, 9), HID, nullptr, nullptr, u1, nullptr, MM, KN, HID, 1);
        // h = (h + (u1 @ node_w2 + nb2)) * node_mask   (in-place per-element on hF)
        launch_gemm(u1, WpN2, LP(L, 10), HID, nmk, hF, h16b, hF, MM, HID, HID, 0);
        float* tmp = cIn; cIn = cOut; cOut = tmp;
    }

    hfinal_kernel<<<MM, 32, 0, stream>>>(hF, emb_out_w, emb_out_b, nmk, hfin);
    geom_kernel<<<BB, NN, 0, stream>>>(cIn, xh, nmk, hfin, (float*)d_out);
}
